// HGT_12051678233155
// MI455X (gfx1250) — compile-verified, matrix operand order checked by reading, not run on hardware
//
#include <hip/hip_runtime.h>
#include <hip/hip_bf16.h>

// ---------------------------------------------------------------------------
// HGT (5-layer) for MI455X / gfx1250, wave32.
//   - Typed projections (K,Q,V,Wa) via v_wmma_f32_16x16x32_bf16 (f32 accum);
//     weights pre-swizzled once per launch into bf16 B-fragment order so the
//     GEMM loop is ds_load_b128 (A) + coalesced global_load_b128 (B) + wmma.
//   - Relation matrices pre-converted to bf16 and staged into LDS with
//     global_load_async_to_lds_b128 (ASYNCcnt) when available.
//   - Edge kernels grid-strided (~4 items/thread) to amortize LDS staging.
//   - Edge softmax: order-preserving-uint atomicMax + f32 atomicAdd;
//     gathered rows prefetched (global_prefetch_b8).
// ---------------------------------------------------------------------------

typedef __attribute__((ext_vector_type(16))) __bf16 v16bf;
typedef __attribute__((ext_vector_type(8)))  float  v8f;
typedef int vi4 __attribute__((vector_size(16)));   // async-LDS builtin arg type

#define DD   128   // feature dim (= H*HD)
#define NH   4     // heads
#define HDIM 32    // head size
#define NTY  3     // node types
#define ETY  6     // edge types
#define NLAY 5

#define RELN (NH * ETY * HDIM * HDIM)               // 24576 elems per layer

// swizzled-weight geometry: [t][ks][ct][lane][16 bf16]
#define WSW_PER_LAYER (NTY * 4 * 8 * 32)            // v16bf elements per layer
#define WSW_ELEMS     (NLAY * WSW_PER_LAYER * 16)   // bf16 elements per tensor

// order-preserving float <-> uint key (for deterministic atomic max)
__device__ __forceinline__ unsigned f2key(float f) {
    unsigned u = __float_as_uint(f);
    return (u & 0x80000000u) ? ~u : (u | 0x80000000u);
}
__device__ __forceinline__ float key2f(unsigned k) {
    unsigned u = (k & 0x80000000u) ? (k ^ 0x80000000u) : ~k;
    return __uint_as_float(u);
}

// Stage `count` bf16 elements (16B-aligned, count%8==0) from global to LDS.
__device__ __forceinline__ void stage_lds_bf16(const __bf16* __restrict__ g,
                                               __bf16* s, int count)
{
#if __has_builtin(__builtin_amdgcn_global_load_async_to_lds_b128)
    for (int i = threadIdx.x * 8; i < count; i += blockDim.x * 8) {
        __builtin_amdgcn_global_load_async_to_lds_b128(
            (vi4*)(g + i), (vi4*)(s + i), /*offset=*/0, /*cpol=*/0);
    }
  #if __has_builtin(__builtin_amdgcn_s_wait_asynccnt)
    __builtin_amdgcn_s_wait_asynccnt(0);
  #else
    asm volatile("s_wait_asynccnt 0" ::: "memory");
  #endif
#else
    const uint4* gs = (const uint4*)g;
    uint4*       ss = (uint4*)s;
    for (int i = threadIdx.x; i < count / 8; i += blockDim.x)
        ss[i] = gs[i];
#endif
    __syncthreads();
}

// ---------------------------------------------------------------------------
// Pre-swizzle f32 weights (L,NT,128,128) into bf16 WMMA B-fragment order:
//   out[((lt*4 + ks)*8 + ct)*32 + lane][j]  with
//   j = 2v+odd, k = ks*32 + ((v<4 ? 2v : 16+2(v-4)) + 8*(lane>>4)) + odd,
//   n = ct*16 + (lane&15)
// ---------------------------------------------------------------------------
__global__ __launch_bounds__(256)
void prep_weights(const float* __restrict__ W, __bf16* __restrict__ out, int total)
{
    int idx = blockIdx.x * blockDim.x + threadIdx.x;
    if (idx >= total) return;
    int j    = idx & 15;
    int lane = (idx >> 4) & 31;
    int ct   = (idx >> 9) & 7;
    int ks   = (idx >> 12) & 3;
    int lt   = idx >> 14;                      // l*NTY + t
    int v    = j >> 1, odd = j & 1;
    int k0   = ((v < 4) ? (2 * v) : (16 + 2 * (v - 4))) + 8 * (lane >> 4);
    int k    = ks * 32 + k0 + odd;
    int n    = ct * 16 + (lane & 15);
    out[idx] = (__bf16)W[((size_t)lt * DD + k) * DD + n];
}

// Plain f32 -> bf16 conversion (relation matrices)
__global__ __launch_bounds__(256)
void f32_to_bf16(const float* __restrict__ in, __bf16* __restrict__ out, int total)
{
    int idx = blockIdx.x * blockDim.x + threadIdx.x;
    if (idx < total) out[idx] = (__bf16)in[idx];
}

// ---------------------------------------------------------------------------
// Typed linear: Y[row] = X[row] @ W[ntype[row]]  (+ optional sigmoid-skip blend)
// Block: 128 threads = 4 waves; block covers 16 rows x 128 cols.
// Wave w computes col tiles {2w, 2w+1}; K = 128 = 4 WMMA steps of 32.
// Computes all 3 types (WMMA) then selects per row -> no sorting needed.
// ---------------------------------------------------------------------------
__global__ __launch_bounds__(128)
void proj_typed(const float* __restrict__ X,      // nrows x 128
                const v16bf* __restrict__ Wsw,    // layer slice, fragment order
                const int*   __restrict__ ntype,  // nrows
                float*       __restrict__ Y,      // nrows x 128
                int nrows, int blend,
                const float* __restrict__ skipv,  // NTY (or null)
                const float* __restrict__ Xres)   // nrows x 128 (or null)
{
    __shared__ __bf16 xs[16 * DD];                // 4 KB A-tile, bf16
    const int tid  = threadIdx.x;
    const int row0 = blockIdx.x * 16;

    for (int i = tid; i < 16 * DD; i += blockDim.x) {
        int r = i >> 7, c = i & 127;
        float f = (row0 + r < nrows) ? X[(size_t)(row0 + r) * DD + c] : 0.0f;
        xs[i] = (__bf16)f;
    }
    __syncthreads();

    const int wave = tid >> 5;
    const int lane = tid & 31;
    const int half = lane >> 4;   // lane group 0-15 / 16-31
    const int mn   = lane & 15;   // M for A-frag, N for B/C/D-frags

    // A fragments for the 4 K-steps (ISA 16-bit 16x32 layout):
    // per lane: bf16[0..7] = K 8*half+0..7, bf16[8..15] = K 16+8*half+0..7
    v16bf afrag[4];
    #pragma unroll
    for (int ks = 0; ks < 4; ++ks) {
        #pragma unroll
        for (int v = 0; v < 8; ++v) {
            int k0 = ((v < 4) ? (2 * v) : (16 + 2 * (v - 4))) + 8 * half;
            afrag[ks][2 * v]     = xs[mn * DD + ks * 32 + k0];
            afrag[ks][2 * v + 1] = xs[mn * DD + ks * 32 + k0 + 1];
        }
    }

    v8f acc[NTY][2];
    #pragma unroll
    for (int t = 0; t < NTY; ++t) {
        #pragma unroll
        for (int ct = 0; ct < 2; ++ct) {
            const int ctile = wave * 2 + ct;
            v8f c = {};
            #pragma unroll
            for (int ks = 0; ks < 4; ++ks) {
                // one contiguous 32B vector load per lane, coalesced per wave
                v16bf b = Wsw[(((t * 4 + ks) * 8) + ctile) * 32 + lane];
                c = __builtin_amdgcn_wmma_f32_16x16x32_bf16(
                        false, afrag[ks], false, b, (short)0, c, false, false);
            }
            acc[t][ct] = c;
        }
    }

    // Epilogue: C/D layout -> element r holds row M = r + 8*half, col = mn.
    #pragma unroll
    for (int r = 0; r < 8; ++r) {
        int row = row0 + r + 8 * half;
        if (row >= nrows) continue;
        int t = ntype[row];
        #pragma unroll
        for (int ct = 0; ct < 2; ++ct) {
            int col = (wave * 2 + ct) * 16 + mn;
            float v = (t == 0) ? acc[0][ct][r]
                    : (t == 1) ? acc[1][ct][r] : acc[2][ct][r];
            if (blend) {
                float alpha = 1.0f / (1.0f + __expf(-skipv[t]));
                v = v * alpha + Xres[(size_t)row * DD + col] * (1.0f - alpha);
            }
            Y[(size_t)row * DD + col] = v;
        }
    }
}

// ---------------------------------------------------------------------------
// Zero/init workspace for one layer: agg (N*128), denom (N*4), amax (N*4).
// ---------------------------------------------------------------------------
__global__ __launch_bounds__(256)
void init_ws(unsigned* __restrict__ amax_key, float* __restrict__ denom,
             float* __restrict__ agg, int nNH, int nND)
{
    int i = blockIdx.x * blockDim.x + threadIdx.x;
    if (i < nND) agg[i] = 0.0f;
    if (i < nNH) { denom[i] = 0.0f; amax_key[i] = f2key(-__builtin_inff()); }
}

// ---------------------------------------------------------------------------
// Pass A: a[e,h] = (att[h,etype] . k[src])·q[dst] * pri / sqrt(HD); atomic max.
// Relation matrices (bf16) async-staged into LDS; grid-stride over (e,h).
// ---------------------------------------------------------------------------
__global__ __launch_bounds__(256)
void edge_scores(const float* __restrict__ kbuf, const float* __restrict__ qbuf,
                 const __bf16* __restrict__ att,  // RELN bf16 (layer slice)
                 const float* __restrict__ pri,   // NH*ETY
                 const int* __restrict__ src, const int* __restrict__ dst,
                 const int* __restrict__ etype,
                 float* __restrict__ a_out, unsigned* __restrict__ amax_key, int nE)
{
    __shared__ __bf16 att_s[RELN];                // 48 KB
    stage_lds_bf16(att, att_s, RELN);

    const int stride = gridDim.x * blockDim.x;
    for (int idx = blockIdx.x * blockDim.x + threadIdx.x; idx < nE * NH;
         idx += stride) {
        int e = idx >> 2, h = idx & 3;
        int s = src[e], d = dst[e], et = etype[e];

        const float* qp = qbuf + (size_t)d * DD + h * HDIM;
        __builtin_prefetch(qp, 0, 1);             // global_prefetch_b8

        const float* kp = kbuf + (size_t)s * DD + h * HDIM;
        float kk[HDIM];
        #pragma unroll
        for (int i = 0; i < HDIM; ++i) kk[i] = kp[i];

        float kw[HDIM];
        #pragma unroll
        for (int o = 0; o < HDIM; ++o) kw[o] = 0.0f;

        const __bf16* ap = att_s + ((h * ETY + et) * HDIM) * HDIM;
        for (int di = 0; di < HDIM; ++di) {
            float kd = kk[di];
            #pragma unroll
            for (int o = 0; o < HDIM; ++o)
                kw[o] += (float)ap[di * HDIM + o] * kd;
        }

        float acc = 0.0f;
        #pragma unroll
        for (int o = 0; o < HDIM; ++o) acc += kw[o] * qp[o];

        float aval = acc * pri[h * ETY + et] * 0.17677669529663687f; // 1/sqrt(32)
        a_out[idx] = aval;
        atomicMax(amax_key + (size_t)d * NH + h, f2key(aval));
    }
}

// ---------------------------------------------------------------------------
// Pass B: denom[dst,h] += exp(a - amax[dst,h])
// ---------------------------------------------------------------------------
__global__ __launch_bounds__(256)
void edge_norm(const float* __restrict__ a_in, const int* __restrict__ dst,
               const unsigned* __restrict__ amax_key, float* __restrict__ denom,
               int nE)
{
    int idx = blockIdx.x * blockDim.x + threadIdx.x;
    if (idx >= nE * NH) return;
    int e = idx >> 2, h = idx & 3;
    int d = dst[e];
    float am = key2f(amax_key[(size_t)d * NH + h]);
    atomicAdd(denom + (size_t)d * NH + h, __expf(a_in[idx] - am));
}

// ---------------------------------------------------------------------------
// Pass C: agg[dst] += (msg[h,etype] . v[src]) * softmax_weight
// ---------------------------------------------------------------------------
__global__ __launch_bounds__(256)
void edge_aggregate(const float* __restrict__ vbuf,
                    const __bf16* __restrict__ msg,  // RELN bf16 (layer slice)
                    const float* __restrict__ a_in,
                    const int* __restrict__ src, const int* __restrict__ dst,
                    const int* __restrict__ etype,
                    const unsigned* __restrict__ amax_key,
                    const float* __restrict__ denom,
                    float* __restrict__ agg, int nE)
{
    __shared__ __bf16 msg_s[RELN];                // 48 KB
    stage_lds_bf16(msg, msg_s, RELN);

    const int stride = gridDim.x * blockDim.x;
    for (int idx = blockIdx.x * blockDim.x + threadIdx.x; idx < nE * NH;
         idx += stride) {
        int e = idx >> 2, h = idx & 3;
        int s = src[e], d = dst[e], et = etype[e];

        const float* vp = vbuf + (size_t)s * DD + h * HDIM;
        __builtin_prefetch(vp, 0, 1);             // global_prefetch_b8

        float am  = key2f(amax_key[(size_t)d * NH + h]);
        float wgt = __expf(a_in[idx] - am) / denom[(size_t)d * NH + h];

        float vv[HDIM];
        #pragma unroll
        for (int i = 0; i < HDIM; ++i) vv[i] = vp[i];

        float mo[HDIM];
        #pragma unroll
        for (int o = 0; o < HDIM; ++o) mo[o] = 0.0f;

        const __bf16* mp = msg_s + ((h * ETY + et) * HDIM) * HDIM;
        for (int di = 0; di < HDIM; ++di) {
            float vd = vv[di];
            #pragma unroll
            for (int o = 0; o < HDIM; ++o)
                mo[o] += (float)mp[di * HDIM + o] * vd;
        }

        float* ag = agg + (size_t)d * DD + h * HDIM;
        #pragma unroll
        for (int o = 0; o < HDIM; ++o)
            atomicAdd(ag + o, mo[o] * wgt);
    }
}

// ---------------------------------------------------------------------------
// Host launcher (graph-capture safe: only kernel launches on `stream`).
// ---------------------------------------------------------------------------
extern "C" void kernel_launch(void* const* d_in, const int* in_sizes, int n_in,
                              void* d_out, int out_size, void* d_ws, size_t ws_size,
                              hipStream_t stream)
{
    const float* x     = (const float*)d_in[0];
    const float* Wk    = (const float*)d_in[1];
    const float* Wq    = (const float*)d_in[2];
    const float* Wv    = (const float*)d_in[3];
    const float* Wa    = (const float*)d_in[4];
    const float* pri   = (const float*)d_in[5];   // (L,H,ET)
    const float* att   = (const float*)d_in[6];   // (L,H,ET,HD,HD)
    const float* msg   = (const float*)d_in[7];   // (L,H,ET,HD,HD)
    const float* skip  = (const float*)d_in[8];   // (L,NT)
    const int*   src   = (const int*)d_in[9];
    const int*   dst   = (const int*)d_in[10];
    const int*   ntype = (const int*)d_in[11];
    const int*   etype = (const int*)d_in[12];

    const int N = in_sizes[11];
    const int E = in_sizes[9];

    // Workspace carve-out (all chunks are multiples of 32 B -> v16bf aligned)
    float* p = (float*)d_ws;
    float* hA    = p; p += (size_t)N * DD;
    float* hB    = p; p += (size_t)N * DD;
    float* kb    = p; p += (size_t)N * DD;
    float* qb    = p; p += (size_t)N * DD;
    float* vb    = p; p += (size_t)N * DD;
    float* agg   = p; p += (size_t)N * DD;
    float* a_e   = p; p += (size_t)E * NH;
    float* denom = p; p += (size_t)N * NH;
    unsigned* amax = (unsigned*)p; p += (size_t)N * NH;
    __bf16* wswK = (__bf16*)p;
    __bf16* wswQ = wswK + WSW_ELEMS;
    __bf16* wswV = wswQ + WSW_ELEMS;
    __bf16* wswA = wswV + WSW_ELEMS;
    __bf16* relA = wswA + WSW_ELEMS;              // bf16 rel_att (L*RELN)
    __bf16* relM = relA + (size_t)NLAY * RELN;    // bf16 rel_msg (L*RELN)

    // Pre-swizzle weights + pre-convert relation matrices (every call)
    const int gbW = (WSW_ELEMS + 255) / 256;
    prep_weights<<<gbW, 256, 0, stream>>>(Wk, wswK, WSW_ELEMS);
    prep_weights<<<gbW, 256, 0, stream>>>(Wq, wswQ, WSW_ELEMS);
    prep_weights<<<gbW, 256, 0, stream>>>(Wv, wswV, WSW_ELEMS);
    prep_weights<<<gbW, 256, 0, stream>>>(Wa, wswA, WSW_ELEMS);
    const int relTot = NLAY * RELN;
    f32_to_bf16<<<(relTot + 255) / 256, 256, 0, stream>>>(att, relA, relTot);
    f32_to_bf16<<<(relTot + 255) / 256, 256, 0, stream>>>(msg, relM, relTot);

    const dim3 gp((N + 15) / 16);                 // proj grid (row tiles)
    const int  nND = N * DD, nNH = N * NH;
    const int  gbN = (nND + 255) / 256;
    const int  gbE  = (E * NH + 255) / 256;                // pass B (no LDS)
    const int  gbE4 = (E * NH + 4 * 256 - 1) / (4 * 256);  // ~4 items/thread

    const float* hin = x;
    for (int l = 0; l < NLAY; ++l) {
        float* hout = (l == NLAY - 1) ? (float*)d_out : ((l & 1) ? hB : hA);
        const v16bf* wk = (const v16bf*)wswK + (size_t)l * WSW_PER_LAYER;
        const v16bf* wq = (const v16bf*)wswQ + (size_t)l * WSW_PER_LAYER;
        const v16bf* wv = (const v16bf*)wswV + (size_t)l * WSW_PER_LAYER;
        const v16bf* wa = (const v16bf*)wswA + (size_t)l * WSW_PER_LAYER;

        proj_typed<<<gp, 128, 0, stream>>>(hin, wk, ntype, kb, N, 0, nullptr, nullptr);
        proj_typed<<<gp, 128, 0, stream>>>(hin, wq, ntype, qb, N, 0, nullptr, nullptr);
        proj_typed<<<gp, 128, 0, stream>>>(hin, wv, ntype, vb, N, 0, nullptr, nullptr);

        init_ws<<<gbN, 256, 0, stream>>>(amax, denom, agg, nNH, nND);

        edge_scores<<<gbE4, 256, 0, stream>>>(kb, qb, relA + (size_t)l * RELN,
                                              pri + (size_t)l * NH * ETY,
                                              src, dst, etype, a_e, amax, E);
        edge_norm<<<gbE, 256, 0, stream>>>(a_e, dst, amax, denom, E);
        edge_aggregate<<<gbE4, 256, 0, stream>>>(vb, relM + (size_t)l * RELN,
                                                 a_e, src, dst, etype, amax,
                                                 denom, agg, E);

        proj_typed<<<gp, 128, 0, stream>>>(agg, wa, ntype, hout, N,
                                           1, skip + (size_t)l * NTY, hin);
        hin = hout;
    }
}
